// RecurrentLinearAttention_53446573031884
// MI455X (gfx1250) — matrix operand order repaired and checked by reading, not compiled, not run on hardware
//
#include <hip/hip_runtime.h>
#include <stdint.h>

#define B_  4
#define S_  4096
#define D_  1024
#define H_  16
#define HD_ 64
#define EPS_ 1e-6f

typedef __attribute__((ext_vector_type(16))) __bf16 v16bf;
typedef __attribute__((ext_vector_type(2)))  __bf16 v2bf;
typedef __attribute__((ext_vector_type(8)))  float  v8f;

union Frag { uint4 q[2]; v16bf v; };

// one float -> bf16 (native convert)
__device__ __forceinline__ uint16_t sbf(float f){
  __bf16 h = (__bf16)f;
  return __builtin_bit_cast(uint16_t, h);
}
// two floats -> packed bf16x2 in a u32
__device__ __forceinline__ uint32_t pack2bf(float a, float b){
#if __has_builtin(__builtin_amdgcn_cvt_pk_bf16_f32)
  v2bf r = __builtin_amdgcn_cvt_pk_bf16_f32(a, b);
  return __builtin_bit_cast(uint32_t, r);
#else
  return (uint32_t)sbf(a) | ((uint32_t)sbf(b) << 16);
#endif
}
__device__ __forceinline__ float bf2f(uint16_t h){
  uint32_t u = ((uint32_t)h) << 16;
  return __builtin_bit_cast(float, u);
}

// A-matrix 16x32 bf16 fragment (ISA 7.12.2): lane m = lane%16, half = lane/16.
// Elements are two contiguous 16B runs: [8*half, +8) and [16+8*half, +8).
// Requires row base 16B-aligned (row stride multiple of 8 elements).
__device__ __forceinline__ v16bf load_frag_a(const uint16_t* row, int half){
  Frag f;
  f.q[0] = *(const uint4*)(row + 8*half);
  f.q[1] = *(const uint4*)(row + 16 + 8*half);
  return f.v;
}
// B-matrix 32x16 bf16 fragment: lane n = lane%16; two 16B runs at elements
// [8*half, +8) and [16+8*half, +8) of the [n][k]-stored operand row.
__device__ __forceinline__ v16bf load_frag_b(const uint16_t* row, int half){
  Frag f;
  f.q[0] = *(const uint4*)(row + 8*half);
  f.q[1] = *(const uint4*)(row + 16 + 8*half);
  return f.v;
}

// ---------------------------------------------------------------------------
// 128x128 bf16-WMMA GEMM, K-step 32.  IN_BF16: A is bf16 (else fp32, converted)
// ACT: apply phi(x)=elu(x)+1 in epilogue.  HEAD_OUT: store bf16 [B,H,S,HD],
// else store fp32 row-major [M,N].
// ---------------------------------------------------------------------------
template<bool IN_BF16, bool ACT, bool HEAD_OUT>
__global__ __launch_bounds__(256) void gemm128(const void* __restrict__ Ap,
                                               const float* __restrict__ W,
                                               void* __restrict__ Outp)
{
  constexpr int N = D_, K = D_;
  constexpr int LDT = 40;                         // 32 + 8 pad: 80B row, 16B-aligned
  __shared__ __align__(16) uint16_t xs[128*LDT];  // A tile, [m][k]
  __shared__ __align__(16) uint16_t wt[128*LDT];  // W tile, transposed [n][k]

  const int tid  = threadIdx.x;
  const int lane = tid & 31, wvid = tid >> 5;
  const int wm = wvid >> 2, wn = wvid & 3;        // 2x4 wave grid
  const int half = lane >> 4, l16 = lane & 15;
  const int m0 = blockIdx.y * 128, n0 = blockIdx.x * 128;

  v8f acc[4][2] = {};

  for (int k0 = 0; k0 < K; k0 += 32){
    if constexpr (!IN_BF16){
      const float* A = (const float*)Ap;
#pragma unroll
      for (int i = 0; i < 4; ++i){
        int f = tid + 256*i;                      // float4 id over 128x32
        int row = f >> 3, c4 = f & 7;
        const float* src = A + (size_t)(m0+row)*K + k0 + c4*4;
        float4 val = *(const float4*)src;
        uint32_t* d = (uint32_t*)&xs[row*LDT + c4*4];
        d[0] = pack2bf(val.x, val.y);
        d[1] = pack2bf(val.z, val.w);
        if (k0 + 32 < K) __builtin_prefetch(src + 32, 0, 1);   // next A tile
      }
    } else {
      const uint16_t* A = (const uint16_t*)Ap;
#pragma unroll
      for (int i = 0; i < 4; ++i){
        int f = tid + 256*i;                      // bf16-quad id over 128x32
        int row = f >> 3, c4 = f & 7;
        const uint16_t* src = A + (size_t)(m0+row)*K + k0 + c4*4;
        *(uint2*)&xs[row*LDT + c4*4] = *(const uint2*)src;
        if (k0 + 32 < K) __builtin_prefetch(src + 32, 0, 1);
      }
    }
#pragma unroll
    for (int i = 0; i < 4; ++i){
      int f = tid + 256*i;                        // float4 id over 32x128
      int kr = f >> 5, nc4 = f & 31;
      const float* src = W + (size_t)(k0+kr)*N + n0 + nc4*4;
      float4 val = *(const float4*)src;
      wt[(nc4*4+0)*LDT + kr] = sbf(val.x);
      wt[(nc4*4+1)*LDT + kr] = sbf(val.y);
      wt[(nc4*4+2)*LDT + kr] = sbf(val.z);
      wt[(nc4*4+3)*LDT + kr] = sbf(val.w);
      if (k0 + 32 < K) __builtin_prefetch(src + (size_t)32*N, 0, 1); // next W tile
    }
    __syncthreads();

    v16bf afr[4], bfr[2];
#pragma unroll
    for (int ms = 0; ms < 4; ++ms)
      afr[ms] = load_frag_a(&xs[(wm*64 + ms*16 + l16)*LDT], half);
#pragma unroll
    for (int ns = 0; ns < 2; ++ns)
      bfr[ns] = load_frag_b(&wt[(wn*32 + ns*16 + l16)*LDT], half);
#pragma unroll
    for (int ms = 0; ms < 4; ++ms)
#pragma unroll
      for (int ns = 0; ns < 2; ++ns)
        acc[ms][ns] = __builtin_amdgcn_wmma_f32_16x16x32_bf16(
            false, afr[ms], false, bfr[ns], (short)0, acc[ms][ns], false, false);
    __syncthreads();
  }

#pragma unroll
  for (int ms = 0; ms < 4; ++ms){
#pragma unroll
    for (int ns = 0; ns < 2; ++ns){
#pragma unroll
      for (int r = 0; r < 8; ++r){
        float v = acc[ms][ns][r];
        if constexpr (ACT) v = (v > 0.f) ? (v + 1.f) : __expf(v);  // elu(x)+1
        int mg = m0 + wm*64 + ms*16 + r + 8*half;
        int ng = n0 + wn*32 + ns*16 + l16;
        if constexpr (HEAD_OUT){
          int b = mg / S_, s = mg % S_;
          int h = ng / HD_, hd = ng % HD_;
          ((uint16_t*)Outp)[(((size_t)(b*H_ + h))*S_ + s)*HD_ + hd] = sbf(v);
        } else {
          ((float*)Outp)[(size_t)mg * N + ng] = v;
        }
      }
    }
  }
}

// ---------------------------------------------------------------------------
// Chunkwise linear-attention scan. One block per (b,h); chunk C=64.
// Per chunk (all via WMMA, fp32 accum):
//   A  = tril(Qc Kc^T)                     den_t = rowsum(A)_t + q_t.z_prev + eps
//   O  = A Vc + Qc S_prev                  out_t = O_t / den_t
//   S += Kc^T Vc    (persistent fp32 C-fragments across chunks)
//   z += colsum(Kc)
// ---------------------------------------------------------------------------
__global__ __launch_bounds__(256) void scan_kernel(const uint16_t* __restrict__ Q,
                                                   const uint16_t* __restrict__ Kf,
                                                   const uint16_t* __restrict__ V,
                                                   uint16_t* __restrict__ O)
{
  constexpr int C  = 64;
  constexpr int NC = S_ / C;
  constexpr int ST = 72;                        // 64 + 8 pad: 144B row, 16B-aligned
  __shared__ __align__(16) uint16_t Qs [64*ST]; // [t][d]
  __shared__ __align__(16) uint16_t Ks [64*ST]; // [t][d]   (B-operand for Q.K^T)
  __shared__ __align__(16) uint16_t KTs[64*ST]; // [d][t]   (A-operand for K^T.V)
  __shared__ __align__(16) uint16_t VTs[64*ST]; // [e][t]   (B-operand, transposed)
  __shared__ __align__(16) uint16_t Ams[64*ST]; // masked A [t][t']
  __shared__ __align__(16) uint16_t STs[64*ST]; // S_prev transposed [e][d]
  __shared__ float zf[64];
  __shared__ float den[64];

  const int tid  = threadIdx.x;
  const int lane = tid & 31, wvid = tid >> 5;
  const int half = lane >> 4, l16 = lane & 15;
  const int tm  = wvid & 3;                     // wave's tile-row (of 4)
  const int tn0 = (wvid >> 2) * 2;              // wave's 2 tile-cols
  const int bh = blockIdx.x;
  const int b = bh / H_, h = bh % H_;
  const size_t base = ((size_t)(b*H_ + h)) * S_ * HD_;

  v8f s_acc[2] = {};                            // persistent S state fragments

  for (int i = tid; i < 64*ST; i += 256) STs[i] = 0;
  if (tid < 64) zf[tid] = 0.f;
  __syncthreads();

  for (int c = 0; c < NC; ++c){
    // ---- phase 0: load chunk (Q,K natural; K,V also transposed) ----
    const size_t cb = base + (size_t)c * C * HD_;
#pragma unroll
    for (int i = 0; i < 4; ++i){
      int p = tid + 256*i;                      // bf16-quad id over 64x64
      int t = p >> 4, dq = (p & 15) * 4;
      uint2 qv = *(const uint2*)(Q  + cb + t*HD_ + dq);
      uint2 kv = *(const uint2*)(Kf + cb + t*HD_ + dq);
      uint2 vv = *(const uint2*)(V  + cb + t*HD_ + dq);
      *(uint2*)&Qs[t*ST + dq] = qv;
      *(uint2*)&Ks[t*ST + dq] = kv;
      KTs[(dq+0)*ST + t] = (uint16_t)kv.x;
      KTs[(dq+1)*ST + t] = (uint16_t)(kv.x >> 16);
      KTs[(dq+2)*ST + t] = (uint16_t)kv.y;
      KTs[(dq+3)*ST + t] = (uint16_t)(kv.y >> 16);
      VTs[(dq+0)*ST + t] = (uint16_t)vv.x;
      VTs[(dq+1)*ST + t] = (uint16_t)(vv.x >> 16);
      VTs[(dq+2)*ST + t] = (uint16_t)vv.y;
      VTs[(dq+3)*ST + t] = (uint16_t)(vv.y >> 16);
    }
    __syncthreads();

    // ---- phase 1: A = Qc.Kc^T ----
    v8f accA[2] = {};
#pragma unroll
    for (int j = 0; j < 2; ++j){
      int tn = tn0 + j;
#pragma unroll
      for (int kh = 0; kh < 2; ++kh){
        v16bf a  = load_frag_a(&Qs[(tm*16 + l16)*ST + 32*kh], half);
        v16bf bb = load_frag_b(&Ks[(tn*16 + l16)*ST + 32*kh], half);
        accA[j] = __builtin_amdgcn_wmma_f32_16x16x32_bf16(
            false, a, false, bb, (short)0, accA[j], false, false);
      }
    }
    // ---- phase 2: causal mask, publish bf16 A ----
#pragma unroll
    for (int j = 0; j < 2; ++j){
      int tn = tn0 + j;
#pragma unroll
      for (int r = 0; r < 8; ++r){
        int t  = tm*16 + r + 8*half;
        int tp = tn*16 + l16;
        float v = (tp <= t) ? accA[j][r] : 0.f;
        Ams[t*ST + tp] = sbf(v);
      }
    }
    __syncthreads();

    // ---- phase 3: denominators (uses z_prev) ----
    if (tid < 64){
      int t = tid;
      float rs = 0.f;
      for (int tp = 0; tp <= t; ++tp) rs += bf2f(Ams[t*ST + tp]);
      float dq = 0.f;
      for (int d = 0; d < 64; ++d) dq += bf2f(Qs[t*ST + d]) * zf[d];
      den[t] = rs + dq + EPS_;
    }

    // ---- phase 4: O = Amsk.Vc + Qc.S_prev ;  S += Kc^T.Vc ----
    v8f accO[2] = {};
#pragma unroll
    for (int j = 0; j < 2; ++j){
      int tn = tn0 + j;
#pragma unroll
      for (int kh = 0; kh < 2; ++kh){
        v16bf a1 = load_frag_a(&Ams[(tm*16 + l16)*ST + 32*kh], half);
        v16bf b1 = load_frag_b(&VTs[(tn*16 + l16)*ST + 32*kh], half);
        accO[j] = __builtin_amdgcn_wmma_f32_16x16x32_bf16(
            false, a1, false, b1, (short)0, accO[j], false, false);
        v16bf a2 = load_frag_a(&Qs[(tm*16 + l16)*ST + 32*kh], half);
        v16bf b2 = load_frag_b(&STs[(tn*16 + l16)*ST + 32*kh], half);
        accO[j] = __builtin_amdgcn_wmma_f32_16x16x32_bf16(
            false, a2, false, b2, (short)0, accO[j], false, false);
        v16bf a3 = load_frag_a(&KTs[(tm*16 + l16)*ST + 32*kh], half);
        s_acc[j] = __builtin_amdgcn_wmma_f32_16x16x32_bf16(
            false, a3, false, b1, (short)0, s_acc[j], false, false);
      }
    }
    __syncthreads();

    // ---- phase 5: publish new S (transposed), z update, write outputs ----
#pragma unroll
    for (int j = 0; j < 2; ++j){
      int tn = tn0 + j;
#pragma unroll
      for (int r = 0; r < 8; ++r){
        int dd = tm*16 + r + 8*half;            // state row (d)
        int e  = tn*16 + l16;                   // state col (e)
        STs[e*ST + dd] = sbf(s_acc[j][r]);
      }
    }
    if (tid < 64){
      int d = tid; float zs = 0.f;
      for (int t = 0; t < 64; ++t) zs += bf2f(Ks[t*ST + d]);
      zf[d] += zs;
    }
#pragma unroll
    for (int j = 0; j < 2; ++j){
      int tn = tn0 + j;
#pragma unroll
      for (int r = 0; r < 8; ++r){
        int t = tm*16 + r + 8*half;
        int e = tn*16 + l16;
        float v = accO[j][r] / den[t];
        int sg = c*C + t;
        O[((size_t)(b*S_ + sg))*D_ + h*HD_ + e] = sbf(v);
      }
    }
    __syncthreads();
  }
}

extern "C" void kernel_launch(void* const* d_in, const int* in_sizes, int n_in,
                              void* d_out, int out_size, void* d_ws, size_t ws_size,
                              hipStream_t stream)
{
  (void)in_sizes; (void)n_in; (void)out_size; (void)ws_size;
  const float* x  = (const float*)d_in[0];
  const float* wq = (const float*)d_in[1];
  const float* wk = (const float*)d_in[2];
  const float* wv = (const float*)d_in[3];
  const float* wo = (const float*)d_in[4];

  const size_t NE = (size_t)B_ * S_ * D_;       // 16.78M elements
  uint16_t* qb = (uint16_t*)d_ws;               // [B,H,S,HD] bf16
  uint16_t* kb = qb + NE;
  uint16_t* vb = kb + NE;
  uint16_t* ab = vb + NE;                       // attn out [B*S, D] bf16

  dim3 blk(256);
  dim3 grd(D_/128, (B_*S_)/128);

  gemm128<false, true,  true ><<<grd, blk, 0, stream>>>(x,  wq, qb); // Q = phi(xWq)
  gemm128<false, true,  true ><<<grd, blk, 0, stream>>>(x,  wk, kb); // K = phi(xWk)
  gemm128<false, false, true ><<<grd, blk, 0, stream>>>(x,  wv, vb); // V = xWv
  scan_kernel<<<dim3(B_*H_), blk, 0, stream>>>(qb, kb, vb, ab);      // chunked scan
  gemm128<true,  false, false><<<grd, blk, 0, stream>>>(ab, wo, (float*)d_out);
}
